// DenseEdgeAwareMPNNConv_48223892799619
// MI455X (gfx1250) — matrix operand order repaired and checked by reading, not compiled, not run on hardware
//
#include <hip/hip_runtime.h>
#include <hip/hip_bf16.h>

typedef __attribute__((ext_vector_type(16))) __bf16 v16bf;
typedef __attribute__((ext_vector_type(8)))  __bf16 v8bf;
typedef __attribute__((ext_vector_type(4)))  __bf16 v4bf;
typedef __attribute__((ext_vector_type(8)))  float  v8f;
typedef __attribute__((ext_vector_type(4)))  float  v4f;

#define NODE_DIM 128
#define EDGE_DIM 32
#define OUT_DIM  128
#define NB       8
#define NN       256
#define NEG_FILL (-1.0e9f)
#define CLAMP_MIN (-1.0e5f)
#define LN_EPS   1e-5f

// ---------------------------------------------------------------------------
// Fragment load helpers: every CDNA5 16-bit A/B fragment a lane needs is two
// contiguous 16-byte runs in our layouts -> 2x ds_load_b128 each.
// ---------------------------------------------------------------------------
__device__ __forceinline__ v16bf ld_frag_split(const __bf16* p, int gap) {
  v8bf lo = *(const v8bf*)p;
  v8bf hi = *(const v8bf*)(p + gap);
  return __builtin_shufflevector(lo, hi, 0, 1, 2, 3, 4, 5, 6, 7,
                                 8, 9, 10, 11, 12, 13, 14, 15);
}
__device__ __forceinline__ v16bf ld_frag_contig(const __bf16* p) {
  v8bf lo = *(const v8bf*)p;
  v8bf hi = *(const v8bf*)(p + 8);
  return __builtin_shufflevector(lo, hi, 0, 1, 2, 3, 4, 5, 6, 7,
                                 8, 9, 10, 11, 12, 13, 14, 15);
}

// ---------------------------------------------------------------------------
// Kernel 1: per-node precompute
//   P[b,n,:] = x @ W1[0:128,:]   + b1    (receiver term, bias folded)
//   Q[b,n,:] = x @ W1[128:256,:]         (sender term)
//   S[b,n,:] = x @ U1_w + U1_b           (skip path for epilogue)
// ---------------------------------------------------------------------------
__global__ __launch_bounds__(128)
void mpnn_precompute(const float* __restrict__ x,
                     const float* __restrict__ W1,
                     const float* __restrict__ b1,
                     const float* __restrict__ U1w,
                     const float* __restrict__ U1b,
                     float* __restrict__ P,
                     float* __restrict__ Q,
                     float* __restrict__ S) {
  __shared__ __align__(16) float xr[NODE_DIM];
  const int row = blockIdx.x;       // b*NN + n
  const int o   = threadIdx.x;      // output column
  xr[o] = x[(size_t)row * NODE_DIM + o];
  __syncthreads();
  float p = 0.f, q = 0.f, s = 0.f;
  #pragma unroll 4
  for (int k = 0; k < NODE_DIM; ++k) {
    const float xv = xr[k];
    p = fmaf(xv, W1[k * OUT_DIM + o], p);
    q = fmaf(xv, W1[(NODE_DIM + k) * OUT_DIM + o], q);
    s = fmaf(xv, U1w[k * OUT_DIM + o], s);
  }
  P[(size_t)row * OUT_DIM + o] = p + b1[o];
  Q[(size_t)row * OUT_DIM + o] = q;
  S[(size_t)row * OUT_DIM + o] = s + U1b[o];
}

// ---------------------------------------------------------------------------
// Kernel 2: convert + swizzle weights into fragment-ready bf16 order.
//   Element (k, n):  ntile = n/16, Nl = n%16, half = (k%32)/16, e = k%16,
//   lane = half*16 + Nl.  Each lane's 16 B-halves become contiguous.
// ---------------------------------------------------------------------------
__global__ void mpnn_convert_weights(const float* __restrict__ W1,
                                     const float* __restrict__ W2,
                                     __bf16* __restrict__ W1c_swz,
                                     __bf16* __restrict__ W2_swz) {
  const int t = blockIdx.x * blockDim.x + threadIdx.x;
  if (t < EDGE_DIM * OUT_DIM) {
    const int k = t >> 7, n = t & 127;
    const int nt = n >> 4, Nl = n & 15;
    const int half = k >> 4, e = k & 15;
    const int lane = half * 16 + Nl;
    W1c_swz[(nt * 32 + lane) * 16 + e] = (__bf16)W1[(2 * NODE_DIM) * OUT_DIM + t];
  }
  if (t < OUT_DIM * OUT_DIM) {
    const int k = t >> 7, n = t & 127;
    const int nt = n >> 4, Nl = n & 15;
    const int ks = k >> 5, kk = k & 31;
    const int half = kk >> 4, e = kk & 15;
    const int lane = half * 16 + Nl;
    W2_swz[(((nt * 4 + ks) * 32) + lane) * 16 + e] = (__bf16)W2[t];
  }
}

// ---------------------------------------------------------------------------
// Kernel 3: fused per-(b,i) message + max-aggregate + epilogue.
//   Software-pipelined: chunk n+1's edge/Q/mask globals are fetched into
//   registers while chunk n's GEMM1 -> LN -> GEMM2 phases run from LDS.
// ---------------------------------------------------------------------------
__global__ __launch_bounds__(128)
void mpnn_main(const float* __restrict__ edge_attr,
               const unsigned char* __restrict__ edge_mask,
               const float* __restrict__ P,
               const float* __restrict__ Q,
               const float* __restrict__ S,
               const __bf16* __restrict__ W1c_swz,
               const __bf16* __restrict__ W2_swz,
               const float* __restrict__ ln1_g,
               const float* __restrict__ ln1_b,
               const float* __restrict__ b2v,
               const float* __restrict__ U2w,
               const float* __restrict__ U2b,
               const float* __restrict__ ln2_g,
               const float* __restrict__ ln2_b,
               float* __restrict__ out) {
  __shared__ __align__(16) __bf16 sW2f[OUT_DIM * OUT_DIM];    // 32 KB swizzled
  __shared__ __align__(16) __bf16 sW1cf[EDGE_DIM * OUT_DIM];  //  8 KB swizzled
  __shared__ __align__(16) __bf16 sA1[16 * EDGE_DIM];         //  1 KB edge tile
  __shared__ __align__(16) float  sPre[16 * OUT_DIM];         //  8 KB P+Q+edge@W1c
  __shared__ __align__(16) __bf16 sA2[16 * OUT_DIM];          //  4 KB relu(LN(h))
  __shared__ __align__(16) float  sPi[OUT_DIM];
  __shared__ __align__(16) float  sAggr[OUT_DIM];
  __shared__ __align__(16) float  sG1[OUT_DIM];
  __shared__ __align__(16) float  sB1[OUT_DIM];
  __shared__ __align__(16) float  sB2[OUT_DIM];
  __shared__ __align__(16) float  sRed1[16][8];
  __shared__ __align__(16) float  sRed2[16][8];
  __shared__ __align__(16) float  sY[OUT_DIM];
  __shared__ unsigned sMaskBits;

  const int tid  = threadIdx.x;       // 0..127
  const int lane = tid & 31;
  const int wave = tid >> 5;          // 0..3
  const int bi   = blockIdx.x;        // b*NN + i
  const int b    = bi >> 8;           // NN == 256

  // stage persistent data (straight vector copies, already swizzled)
  {
    const int4* src2 = (const int4*)W2_swz;  // 2048 int4
    int4* dst2 = (int4*)sW2f;
    for (int t = tid; t < (OUT_DIM * OUT_DIM) / 8; t += 128) dst2[t] = src2[t];
    const int4* src1 = (const int4*)W1c_swz; // 512 int4
    int4* dst1 = (int4*)sW1cf;
    for (int t = tid; t < (EDGE_DIM * OUT_DIM) / 8; t += 128) dst1[t] = src1[t];
  }
  sPi[tid]   = P[(size_t)bi * OUT_DIM + tid];
  sAggr[tid] = NEG_FILL;
  sG1[tid]   = ln1_g[tid];
  sB1[tid]   = ln1_b[tid];
  sB2[tid]   = b2v[tid];
  __syncthreads();

  // CDNA5 WMMA wave32 fragment geometry
  const int Nl      = lane & 15;      // output column within 16-wide N tile
  const int halfsel = lane >> 4;      // lane half
  const int Mbase   = halfsel * 8;    // C/D rows owned by this lane
  const int Ml      = lane & 15;      // A-matrix row
  const int aOff    = halfsel * 16;   // A K-group offset for this lane half

  const float* ebase = edge_attr + (size_t)bi * NN * EDGE_DIM;
  const unsigned char* mbase = edge_mask + (size_t)bi * NN;
  const v4f* Qall4 = (const v4f*)(Q + ((size_t)b * NN) * OUT_DIM);
  const v4f* Pi4   = (const v4f*)sPi;

  // ---- pipeline registers + prologue loads (chunk 0) ---------------------
  v4f evR;
  v4f qvR[4];
  unsigned char mvR = 0;
  {
    evR = ((const v4f*)ebase)[tid];
    #pragma unroll
    for (int p = 0; p < 4; ++p) {
      const int f = tid + p * 128;
      qvR[p] = Qall4[(size_t)(f >> 5) * 32 + (f & 31)];
    }
    if (wave == 0 && lane < 16) mvR = mbase[lane];
  }

  for (int j0 = 0; j0 < NN; j0 += 16) {
    // ---- commit staged chunk to LDS ------------------------------------
    {
      v4bf eb;
      eb[0] = (__bf16)evR[0]; eb[1] = (__bf16)evR[1];
      eb[2] = (__bf16)evR[2]; eb[3] = (__bf16)evR[3];
      *(v4bf*)(sA1 + (tid >> 3) * EDGE_DIM + (tid & 7) * 4) = eb;

      #pragma unroll
      for (int p = 0; p < 4; ++p) {
        const int f  = tid + p * 128;
        const int c4 = f & 31;
        v4f qv = qvR[p];
        const v4f pv = Pi4[c4];
        qv[0] += pv[0]; qv[1] += pv[1]; qv[2] += pv[2]; qv[3] += pv[3];
        ((v4f*)sPre)[f] = qv;
      }
      if (wave == 0) {
        const unsigned bits = (unsigned)__ballot(mvR != 0) & 0xFFFFu;
        if (lane == 0) sMaskBits = bits;
      }
    }
    __syncthreads();

    // ---- issue next chunk's global loads (latency hidden by compute) ---
    if (j0 + 16 < NN) {
      const int jn = j0 + 16;
      evR = ((const v4f*)ebase)[jn * 8 + tid];
      #pragma unroll
      for (int p = 0; p < 4; ++p) {
        const int f = tid + p * 128;
        qvR[p] = Qall4[(size_t)(jn + (f >> 5)) * 32 + (f & 31)];
      }
      if (wave == 0 && lane < 16) mvR = mbase[jn + lane];
      if (jn + 16 < NN)
        __builtin_prefetch(ebase + (size_t)(jn + 16) * EDGE_DIM + tid * 4, 0, 1);
    }

    // ---- GEMM1: edge[16x32] @ W1c[32x128], accumulate into sPre --------
    {
      const v16bf a = ld_frag_split(sA1 + Ml * EDGE_DIM + aOff, 8);
      #pragma unroll
      for (int nt = 0; nt < 2; ++nt) {
        const int ntg = wave * 2 + nt;
        const v16bf bb = ld_frag_contig(sW1cf + (ntg * 32 + lane) * 16);
        v8f c = {};
        c = __builtin_amdgcn_wmma_f32_16x16x32_bf16(
                false, a, false, bb, (short)0, c, false, false);
        const int col = ntg * 16 + Nl;
        #pragma unroll
        for (int r = 0; r < 8; ++r)
          atomicAdd(&sPre[(Mbase + r) * OUT_DIM + col], c[r]);  // ds_add_f32
      }
    }
    __syncthreads();

    // ---- LayerNorm + ReLU per row -> bf16 A of GEMM2 -------------------
    {
      const int r   = tid >> 3;   // row 0..15
      const int seg = tid & 7;    // 16-col segment
      const v4f* row4 = (const v4f*)(sPre + r * OUT_DIM + seg * 16);
      v4f vals[4];
      float s1 = 0.f, s2 = 0.f;
      #pragma unroll
      for (int q = 0; q < 4; ++q) {
        vals[q] = row4[q];
        #pragma unroll
        for (int u = 0; u < 4; ++u) { s1 += vals[q][u]; s2 += vals[q][u] * vals[q][u]; }
      }
      sRed1[r][seg] = s1;
      sRed2[r][seg] = s2;
      __syncthreads();
      float t1 = 0.f, t2 = 0.f;
      #pragma unroll
      for (int ss = 0; ss < 8; ++ss) { t1 += sRed1[r][ss]; t2 += sRed2[r][ss]; }
      const float mean = t1 * (1.0f / OUT_DIM);
      const float var  = t2 * (1.0f / OUT_DIM) - mean * mean;
      const float inv  = rsqrtf(var + LN_EPS);
      #pragma unroll
      for (int q = 0; q < 4; ++q) {
        const v4f g4 = ((const v4f*)sG1)[seg * 4 + q];
        const v4f b4 = ((const v4f*)sB1)[seg * 4 + q];
        v4bf ob;
        #pragma unroll
        for (int u = 0; u < 4; ++u) {
          const float v = (vals[q][u] - mean) * (g4[u] * inv) + b4[u];
          ob[u] = (__bf16)fmaxf(v, 0.f);
        }
        *(v4bf*)(sA2 + r * OUT_DIM + seg * 16 + q * 4) = ob;
      }
    }
    __syncthreads();

    // ---- GEMM2: act[16x128] @ W2[128x128] + b2, mask, max-reduce -------
    {
      const unsigned mbits = sMaskBits;
      #pragma unroll
      for (int nt = 0; nt < 2; ++nt) {
        const int ntg = wave * 2 + nt;
        v8f c = {};
        #pragma unroll
        for (int ks = 0; ks < 4; ++ks) {
          const v16bf a  = ld_frag_split(sA2 + Ml * OUT_DIM + ks * 32 + aOff, 8);
          const v16bf bb = ld_frag_contig(sW2f + ((ntg * 4 + ks) * 32 + lane) * 16);
          c = __builtin_amdgcn_wmma_f32_16x16x32_bf16(
                  false, a, false, bb, (short)0, c, false, false);
        }
        const int col  = ntg * 16 + Nl;
        const float bias = sB2[col];
        float mx = NEG_FILL;
        #pragma unroll
        for (int r = 0; r < 8; ++r) {
          const float v = c[r] + bias;
          mx = fmaxf(mx, ((mbits >> (Mbase + r)) & 1u) ? v : NEG_FILL);
        }
        // fold the two 8-row halves (lane L <-> L^16 share a column)
        mx = fmaxf(mx, __shfl_xor(mx, 16, 32));
        if (halfsel == 0) sAggr[col] = fmaxf(sAggr[col], mx);
      }
    }
    __syncthreads();
  }

  // ---- epilogue: clamp, aggr@U2, skip add, LN2, ReLU ---------------------
  const float aggr_c = fmaxf(sAggr[tid], CLAMP_MIN);
  __syncthreads();
  sAggr[tid] = aggr_c;
  __syncthreads();

  {
    const int o = tid;
    float acc = 0.f;
    #pragma unroll 4
    for (int k = 0; k < OUT_DIM; ++k)
      acc = fmaf(sAggr[k], U2w[k * OUT_DIM + o], acc);
    const float y = S[(size_t)bi * OUT_DIM + o] + acc + U2b[o];
    sY[o] = y;
    __syncthreads();
    float s1 = 0.f, s2 = 0.f;
    const v4f* y4 = (const v4f*)sY;
    #pragma unroll
    for (int k = 0; k < OUT_DIM / 4; ++k) {
      const v4f v = y4[k];
      #pragma unroll
      for (int u = 0; u < 4; ++u) { s1 += v[u]; s2 += v[u] * v[u]; }
    }
    const float mean = s1 * (1.0f / OUT_DIM);
    const float var  = s2 * (1.0f / OUT_DIM) - mean * mean;
    const float inv  = rsqrtf(var + LN_EPS);
    const float r    = (y - mean) * (ln2_g[o] * inv) + ln2_b[o];
    out[(size_t)bi * OUT_DIM + o] = fmaxf(r, 0.f);
  }
}

// ---------------------------------------------------------------------------
extern "C" void kernel_launch(void* const* d_in, const int* in_sizes, int n_in,
                              void* d_out, int out_size, void* d_ws, size_t ws_size,
                              hipStream_t stream) {
  const float* x          = (const float*)d_in[0];
  const float* edge_attr  = (const float*)d_in[1];
  const unsigned char* em = (const unsigned char*)d_in[2];
  const float* W1         = (const float*)d_in[3];
  const float* b1         = (const float*)d_in[4];
  const float* ln1_g      = (const float*)d_in[5];
  const float* ln1_b      = (const float*)d_in[6];
  const float* W2         = (const float*)d_in[7];
  const float* b2         = (const float*)d_in[8];
  const float* U1_w       = (const float*)d_in[9];
  const float* U1_b       = (const float*)d_in[10];
  const float* U2_w       = (const float*)d_in[11];
  const float* U2_b       = (const float*)d_in[12];
  const float* ln2_g      = (const float*)d_in[13];
  const float* ln2_b      = (const float*)d_in[14];
  float* out              = (float*)d_out;

  char* ws = (char*)d_ws;
  float* P = (float*)ws;        ws += (size_t)NB * NN * OUT_DIM * sizeof(float);
  float* Q = (float*)ws;        ws += (size_t)NB * NN * OUT_DIM * sizeof(float);
  float* S = (float*)ws;        ws += (size_t)NB * NN * OUT_DIM * sizeof(float);
  __bf16* W1c_swz = (__bf16*)ws; ws += (size_t)EDGE_DIM * OUT_DIM * sizeof(__bf16);
  __bf16* W2_swz  = (__bf16*)ws; ws += (size_t)OUT_DIM * OUT_DIM * sizeof(__bf16);

  mpnn_precompute<<<NB * NN, 128, 0, stream>>>(x, W1, b1, U1_w, U1_b, P, Q, S);
  mpnn_convert_weights<<<(OUT_DIM * OUT_DIM + 255) / 256, 256, 0, stream>>>(
      W1, W2, W1c_swz, W2_swz);
  mpnn_main<<<NB * NN, 128, 0, stream>>>(edge_attr, em, P, Q, S,
                                         W1c_swz, W2_swz,
                                         ln1_g, ln1_b, b2, U2_w, U2_b,
                                         ln2_g, ln2_b, out);
}
